// MeshfreeKAN2D_12996571038259
// MI455X (gfx1250) — matrix operand order repaired and checked by reading, not compiled, not run on hardware
//
#include <hip/hip_runtime.h>

typedef __attribute__((ext_vector_type(16))) _Float16 v16h;
typedef __attribute__((ext_vector_type(8)))  float    v8f;

#define MAXN 2048
#define SUPPORT_RADIUS 0.1f

__device__ __forceinline__ float hat1(float v, float g) {
    // relu(1 - |v - g| / 0.5)
    return fmaxf(0.0f, 1.0f - fabsf(v - g) * 2.0f);
}

__device__ __forceinline__ float cubic_window(float dist) {
    float q = dist * 10.0f;   // dist / 0.1
    float inner = 2.0f / 3.0f + q * q * (4.0f * q - 4.0f);
    float outer = 4.0f / 3.0f + q * (-4.0f + q * (4.0f - (4.0f / 3.0f) * q));
    float w = (q <= 0.5f) ? inner : outer;
    return (q <= 1.0f) ? w : 0.0f;
}

__device__ __forceinline__ float softplus_f(float x) {
    return fmaxf(x, 0.0f) + log1pf(expf(-fabsf(x)));
}

__global__ __launch_bounds__(128) void meshfree_kan_kernel(
    const float* __restrict__ x,     // [B,2]
    const float* __restrict__ nodes, // [N,2]
    const float* __restrict__ W1a,   // [16,7]
    const float* __restrict__ W1b,   // [16,7]
    const float* __restrict__ W2,    // [1,112]
    const float* __restrict__ w,     // [N,1]
    float* __restrict__ out,         // [B,1]
    int B, int N)
{
    __shared__ float s_nodes[MAXN * 2];
    __shared__ float s_w[MAXN];

    const int tid  = threadIdx.x;
    const int lane = tid & 31;
    const int warp = tid >> 5;

    // stage nodes + w into LDS once per workgroup (24 KB)
    for (int i = tid; i < N * 2; i += (int)blockDim.x) s_nodes[i] = nodes[i];
    for (int i = tid; i < N;     i += (int)blockDim.x) s_w[i]     = w[i];
    __syncthreads();

    const int b = blockIdx.x * 4 + warp;   // one wave per query row
    if (b >= B) return;

    const int  u  = lane & 15;       // hidden unit index (B-matrix column / D N-index)
    const bool up = lane >= 16;      // upper half of wave (A-matrix K=8..15 slice)

    // ---- Build B matrix once: [W1a(7) ; W1b(7) ; 0-pad] as 32x16 f16 ----
    // B layout (16-bit, 32x16): element e<8 maps to K = e + 8*(lane>=16),
    // element e>=8 maps to K = 16 + 8*(lane>=16) + (e-8), column N = lane&15.
    v16h bm;
#pragma unroll
    for (int e = 0; e < 16; ++e) bm[e] = (_Float16)0.0f;
    if (lane < 16) {
#pragma unroll
        for (int e = 0; e < 7; ++e) bm[e]     = (_Float16)W1a[lane * 7 + e];
        bm[7] = (_Float16)W1b[lane * 7 + 0];
    } else {
#pragma unroll
        for (int e = 0; e < 6; ++e) bm[e] = (_Float16)W1b[u * 7 + 1 + e];
    }

    // W2 row for this lane's hidden unit
    float w2r[7];
#pragma unroll
    for (int i = 0; i < 7; ++i) w2r[i] = W2[u * 7 + i];

    const float qx = x[2 * b + 0];
    const float qy = x[2 * b + 1];

    float num = 0.0f, den = 0.0f;
    const int pa = lane & 15;   // A-matrix row (pair) this lane packs data for

    for (int base = 0; base < N; base += 16) {
        int na  = base + pa;
        int nac = (na < N) ? na : 0;
        float nx = s_nodes[2 * nac + 0];
        float ny = s_nodes[2 * nac + 1];
        float dx = qx - nx, dy = qy - ny;
        if (na >= N) { dx = 1.0e3f; dy = 1.0e3f; }
        float dist = sqrtf(dx * dx + dy * dy);

        // window == 0 for every pair in tile -> exact-zero contribution, skip (uniform branch)
        if (__ballot(dist <= SUPPORT_RADIUS) == 0ull) continue;

        // ---- Layer 1 hats ----
        float vx = dx * 10.0f, vy = dy * 10.0f;
        float hx[7], hy[7];
#pragma unroll
        for (int i = 0; i < 7; ++i) {
            float g = -1.5f + 0.5f * (float)i;
            hx[i] = hat1(vx, g);
            hy[i] = hat1(vy, g);
        }

        // ---- Pack A (16x32 f16): K = [b0(7), b1(7), zeros...] ----
        // lane<16: elements 0..7 = K0..7 ; lane>=16: elements 0..7 = K8..15
        v16h am;
        am[0] = (_Float16)(up ? hy[1] : hx[0]);
        am[1] = (_Float16)(up ? hy[2] : hx[1]);
        am[2] = (_Float16)(up ? hy[3] : hx[2]);
        am[3] = (_Float16)(up ? hy[4] : hx[3]);
        am[4] = (_Float16)(up ? hy[5] : hx[4]);
        am[5] = (_Float16)(up ? hy[6] : hx[5]);
        am[6] = (_Float16)(up ? 0.0f  : hx[6]);
        am[7] = (_Float16)(up ? 0.0f  : hy[0]);
#pragma unroll
        for (int e = 8; e < 16; ++e) am[e] = (_Float16)0.0f;

        v8f c;
#pragma unroll
        for (int i = 0; i < 8; ++i) c[i] = 0.0f;

        // hidden = A x B : one matrix op for 16 pairs x 16 units
        v8f d = __builtin_amdgcn_wmma_f32_16x16x32_f16(
            false, am, false, bm, (short)0, c, false, false);
        // D layout: lane l, vgpr v -> hidden[pair = v + 8*(l>=16)][unit = l&15]

        // ---- Layer 2: per (pair,unit) hat-dot with W2 row, f32 ----
        float s[8];
#pragma unroll
        for (int v = 0; v < 8; ++v) {
            float h   = d[v];
            float acc = 0.0f;
#pragma unroll
            for (int i = 0; i < 7; ++i) {
                float g = -1.5f + 0.5f * (float)i;
                acc = fmaf(w2r[i], hat1(h, g), acc);
            }
            s[v] = acc;
        }

        // reduce over 16 units: butterfly within each 16-lane half
#pragma unroll
        for (int m = 1; m < 16; m <<= 1) {
#pragma unroll
            for (int v = 0; v < 8; ++v) s[v] += __shfl_xor(s[v], m, 32);
        }

        // select this lane's pair result
        int sub = lane & 7;
        float pre = s[0];
#pragma unroll
        for (int v = 1; v < 8; ++v) pre = (sub == v) ? s[v] : pre;

        int p  = sub + (up ? 8 : 0);
        int np = base + p;
        bool act = ((lane & 15) < 8) && (np < N);
        int npc = (np < N) ? np : 0;

        float nx2 = s_nodes[2 * npc + 0];
        float ny2 = s_nodes[2 * npc + 1];
        float dx2 = qx - nx2, dy2 = qy - ny2;
        float d2  = sqrtf(dx2 * dx2 + dy2 * dy2);
        float wnd = cubic_window(d2);
        float phi = softplus_f(pre) * wnd;

        num += act ? phi * s_w[npc] : 0.0f;
        den += act ? phi : 0.0f;
    }

    // full-wave reduction of num/den
#pragma unroll
    for (int m = 1; m < 32; m <<= 1) {
        num += __shfl_xor(num, m, 32);
        den += __shfl_xor(den, m, 32);
    }

    if (lane == 0) {
        float res;
        if (fabsf(den) >= 1e-14f) {
            res = num / (den + 1e-12f);
        } else {
            // kNN fallback (statistically never taken; serial over LDS)
            const int K = (N < 8) ? N : 8;
            float bd[8]; int bi[8];
#pragma unroll
            for (int k = 0; k < 8; ++k) { bd[k] = 3.4e38f; bi[k] = 0; }
            for (int n = 0; n < N; ++n) {
                float ddx = qx - s_nodes[2 * n + 0];
                float ddy = qy - s_nodes[2 * n + 1];
                float dn = sqrtf(ddx * ddx + ddy * ddy);
                if (dn < bd[K - 1]) {
                    int k = K - 1;
                    while (k > 0 && bd[k - 1] > dn) {
                        bd[k] = bd[k - 1]; bi[k] = bi[k - 1]; --k;
                    }
                    bd[k] = dn; bi[k] = n;
                }
            }
            float alpha = 20.0f / SUPPORT_RADIUS;  // 200
            float sw = 0.0f;
            for (int k = 0; k < K; ++k) sw += expf(-alpha * bd[k]);
            sw += 1e-18f;
            float acc = 0.0f;
            for (int k = 0; k < K; ++k)
                acc += (expf(-alpha * bd[k]) / sw) * s_w[bi[k]];
            res = acc;
        }
        out[b] = res;
    }
}

extern "C" void kernel_launch(void* const* d_in, const int* in_sizes, int n_in,
                              void* d_out, int out_size, void* d_ws, size_t ws_size,
                              hipStream_t stream) {
    const float* x     = (const float*)d_in[0];
    const float* nodes = (const float*)d_in[1];
    const float* W1a   = (const float*)d_in[2];
    const float* W1b   = (const float*)d_in[3];
    const float* W2    = (const float*)d_in[4];
    const float* w     = (const float*)d_in[5];
    float* out = (float*)d_out;

    int B = in_sizes[0] / 2;
    int N = in_sizes[1] / 2;
    if (N > MAXN) N = MAXN;   // LDS staging capacity guard

    const int waves_per_block = 4;
    dim3 block(32 * waves_per_block);
    dim3 grid((B + waves_per_block - 1) / waves_per_block);
    meshfree_kan_kernel<<<grid, block, 0, stream>>>(x, nodes, W1a, W1b, W2, w, out, B, N);
}
